// CausalSelfAttention_47785806135455
// MI455X (gfx1250) — matrix operand order repaired and checked by reading, not compiled, not run on hardware
//
#include <hip/hip_runtime.h>
#include <hip/hip_bf16.h>

// ---------------------------------------------------------------------------
// Causal self-attention, bf16 WMMA (gfx1250), fp32 accumulate.
// Double-buffered async global->LDS tile pipeline everywhere.
// ---------------------------------------------------------------------------

typedef __attribute__((ext_vector_type(16))) __bf16 v16bf;
typedef __attribute__((ext_vector_type(8)))  float  v8f;

#define EMBED   1024
#define HEADS   16
#define HDIM    64
#define BATCH   4
#define SEQ     2048
#define MTOT    (BATCH * SEQ)     /* 8192 */
#define N_QKV   (3 * EMBED)       /* 3072 */

__device__ __forceinline__ unsigned short f2bf(float f) {
  unsigned int u = __float_as_uint(f);
  unsigned int r = (u + 0x7FFFu + ((u >> 16) & 1u)) >> 16;   // RNE
  return (unsigned short)r;
}

union FragBF { uint4 u[2]; v16bf v; };

// A-style fragment: 8 bf16 at [idx0], 8 bf16 at [idx0+16]  (lane base = (lane>>4)*8)
__device__ __forceinline__ v16bf frag_split(const unsigned short* p, int idx0) {
  FragBF f;
  const uint4* q = reinterpret_cast<const uint4*>(p + idx0);
  f.u[0] = q[0];
  f.u[1] = q[2];
  return f.v;
}
// B-style fragment: 16 contiguous bf16 at [idx0]  (lane base = (lane>>4)*16)
__device__ __forceinline__ v16bf frag_contig(const unsigned short* p, int idx0) {
  FragBF f;
  const uint4* q = reinterpret_cast<const uint4*>(p + idx0);
  f.u[0] = q[0];
  f.u[1] = q[1];
  return f.v;
}

// CDNA5 async global->LDS copy (ASYNCcnt-tracked, no VGPR data path).
// LDS byte address = low 32 bits of the flat pointer (ISA 10.2 aperture rule).
__device__ __forceinline__ void async_ld_b128(const unsigned short* lds_dst,
                                              const unsigned short* gsrc) {
  unsigned lds = (unsigned)(unsigned long long)(uintptr_t)lds_dst;
  unsigned long long ga = (unsigned long long)(uintptr_t)gsrc;
  asm volatile("global_load_async_to_lds_b128 %0, %1, off"
               :: "v"(lds), "v"(ga) : "memory");
}
__device__ __forceinline__ void wait_async0() {
  asm volatile("s_wait_asynccnt 0x0" ::: "memory");
}

// ---------------------------------------------------------------------------
// fp32 -> bf16 conversion (x)
// ---------------------------------------------------------------------------
__global__ void cvt_f32_bf16_kernel(const float* __restrict__ in,
                                    unsigned short* __restrict__ out, int n) {
  int i = blockIdx.x * blockDim.x + threadIdx.x;
  int stride = gridDim.x * blockDim.x;
  for (; i < n; i += stride) out[i] = f2bf(in[i]);
}

// ---------------------------------------------------------------------------
// fused fp32 -> bf16 convert + transpose: in [K][N] f32 -> out [N][K] bf16
// ---------------------------------------------------------------------------
__global__ __launch_bounds__(256)
void cvt_transpose_kernel(const float* __restrict__ in,
                          unsigned short* __restrict__ out, int K, int N) {
  __shared__ unsigned short tile[32][33];
  const int kt = blockIdx.y * 32;
  const int nt = blockIdx.x * 32;
  const int tx = threadIdx.x & 31;
  const int ty = threadIdx.x >> 5;      // 0..7
#pragma unroll
  for (int i = 0; i < 32; i += 8)
    tile[ty + i][tx] = f2bf(in[(size_t)(kt + ty + i) * N + nt + tx]);
  __syncthreads();
#pragma unroll
  for (int i = 0; i < 32; i += 8)
    out[(size_t)(nt + ty + i) * K + kt + tx] = tile[tx][ty + i];
}

// ---------------------------------------------------------------------------
// GEMM core tile fill: async copy 128x64 bf16 A and B tiles into LDS buffers.
// ---------------------------------------------------------------------------
__device__ __forceinline__ void gemm_fill_async(unsigned short* Abuf, unsigned short* Bbuf,
                                                const unsigned short* __restrict__ Ag,
                                                const unsigned short* __restrict__ Bg,
                                                int m0, int n0, int k0, int tid) {
#pragma unroll
  for (int j = 0; j < 4; ++j) {                  // 1024 uint4 per tile
    int idx = j * 256 + tid;
    int row = idx >> 3;                          // 8 uint4 per 64-el row
    int c8  = (idx & 7) * 8;
    async_ld_b128(&Abuf[row * 64 + c8], &Ag[(size_t)(m0 + row) * EMBED + k0 + c8]);
    async_ld_b128(&Bbuf[row * 64 + c8], &Bg[(size_t)(n0 + row) * EMBED + k0 + c8]);
  }
}

// ---------------------------------------------------------------------------
// QKV GEMM: X[MTOT,EMBED] x Wt[N_QKV,EMBED]^T + bias -> Q/K bf16 [B,H,T,D],
// V bf16 pre-transposed [B,H,D,T].  Block 128x128, 8 waves 32x64, k-step 64,
// double-buffered async LDS pipeline.
// ---------------------------------------------------------------------------
__global__ __launch_bounds__(256)
void qkv_gemm_kernel(const unsigned short* __restrict__ Xb,
                     const unsigned short* __restrict__ WbT,   // [N_QKV][EMBED]
                     const float* __restrict__ bias,
                     unsigned short* __restrict__ Qb,
                     unsigned short* __restrict__ Kb,
                     unsigned short* __restrict__ Vtb) {
  __shared__ __align__(16) unsigned short As[2][128 * 64];   // [m][k]
  __shared__ __align__(16) unsigned short Bs[2][128 * 64];   // [n][k]
  const int tid  = threadIdx.x;
  const int lane = tid & 31;
  const int wave = tid >> 5;
  const int m0 = blockIdx.y * 128;
  const int n0 = blockIdx.x * 128;
  const int waveM = (wave >> 1) * 32;
  const int waveN = (wave & 1) * 64;
  const int nIn = lane & 15;
  const int half = lane >> 4;

  v8f acc[2][4];
#pragma unroll
  for (int mi = 0; mi < 2; ++mi)
#pragma unroll
    for (int ni = 0; ni < 4; ++ni) acc[mi][ni] = (v8f){};

  const int NIT = EMBED / 64;
  gemm_fill_async(As[0], Bs[0], Xb, WbT, m0, n0, 0, tid);
  wait_async0();
  __syncthreads();

  for (int it = 0; it < NIT; ++it) {
    const int cur = it & 1;
    if (it + 1 < NIT)
      gemm_fill_async(As[cur ^ 1], Bs[cur ^ 1], Xb, WbT, m0, n0, (it + 1) * 64, tid);

    const unsigned short* Ac = As[cur];
    const unsigned short* Bc = Bs[cur];
    v16bf af[2][2];
#pragma unroll
    for (int mi = 0; mi < 2; ++mi)
#pragma unroll
      for (int kk = 0; kk < 2; ++kk)
        af[mi][kk] = frag_split(Ac, (waveM + mi * 16 + nIn) * 64 + kk * 32 + half * 8);
#pragma unroll
    for (int ni = 0; ni < 4; ++ni) {
#pragma unroll
      for (int kk = 0; kk < 2; ++kk) {
        v16bf bf = frag_split(Bc, (waveN + ni * 16 + nIn) * 64 + kk * 32 + half * 8);
#pragma unroll
        for (int mi = 0; mi < 2; ++mi)
          acc[mi][ni] = __builtin_amdgcn_wmma_f32_16x16x32_bf16(
              false, af[mi][kk], false, bf, (short)0, acc[mi][ni], false, false);
      }
    }
    wait_async0();
    __syncthreads();
  }

  // epilogue: bias + scatter. Q,K -> [B,H,T,D]; V -> [B,H,D,T] (transposed)
#pragma unroll
  for (int mi = 0; mi < 2; ++mi)
#pragma unroll
    for (int ni = 0; ni < 4; ++ni) {
      int ncol = n0 + waveN + ni * 16 + nIn;
      float bv = bias[ncol];
      int sel = ncol >> 10;
      int rem = ncol & 1023;
      int h = rem >> 6, d = rem & 63;
#pragma unroll
      for (int r = 0; r < 8; ++r) {
        int mrow = m0 + waveM + mi * 16 + half * 8 + r;
        int bidx = mrow >> 11;            // / SEQ
        int t    = mrow & (SEQ - 1);
        unsigned short val = f2bf(acc[mi][ni][r] + bv);
        if (sel == 2)
          Vtb[((size_t)(bidx * HEADS + h) * HDIM + d) * SEQ + t] = val;
        else if (sel == 1)
          Kb[((size_t)(bidx * HEADS + h) * SEQ + t) * HDIM + d] = val;
        else
          Qb[((size_t)(bidx * HEADS + h) * SEQ + t) * HDIM + d] = val;
      }
    }
}

// ---------------------------------------------------------------------------
// Flash attention: grid (T/128, B*H), 8 waves, 16 q-rows/wave, 64-key blocks.
// K tile natural [t][d]; V tile already transposed in global [d][t].
// Double-buffered async K/V tile pipeline.
// ---------------------------------------------------------------------------
__global__ __launch_bounds__(256)
void attn_kernel(const unsigned short* __restrict__ Qb,
                 const unsigned short* __restrict__ Kb,
                 const unsigned short* __restrict__ Vtb,
                 unsigned short* __restrict__ Ab) {
  __shared__ __align__(16) unsigned short Ks[2][64 * 64];    // [t][d]
  __shared__ __align__(16) unsigned short Vt[2][64 * 64];    // [d][t]
  __shared__ __align__(16) unsigned short Ps[8 * 16 * 64];   // per-wave P tile
  const int tid  = threadIdx.x;
  const int lane = tid & 31;
  const int wave = tid >> 5;
  const int bh = blockIdx.y;
  const int q0 = blockIdx.x * 128;
  const int qrow0 = q0 + wave * 16;
  const int half = lane >> 4;
  const int nIn  = lane & 15;
  const float scale = 0.125f;      // 1/sqrt(64)

  v16bf aq[2];
  {
    const unsigned short* qptr = Qb + (size_t)(bh * SEQ + qrow0 + nIn) * HDIM;
#pragma unroll
    for (int kk = 0; kk < 2; ++kk)
      aq[kk] = frag_split(qptr, kk * 32 + half * 8);
  }

  v8f o[4];
#pragma unroll
  for (int ni = 0; ni < 4; ++ni) o[ni] = (v8f){};
  float mrow[8], lrow[8];
#pragma unroll
  for (int r = 0; r < 8; ++r) { mrow[r] = -1e30f; lrow[r] = 0.f; }

  // per-thread fill coordinates (512 uint4 per tile, 2 per thread)
  const int frow0 = tid >> 2;            // j=0 row
  const int fc80  = (tid & 3) * 16;      // two c8 slots: fc80, fc80+8
  auto fill_kv = [&](int kv0, int buf) {
#pragma unroll
    for (int j = 0; j < 2; ++j) {
      int row = frow0 + (j ? 0 : 0);     // rows 0..63 from tid>>2
      int c8  = fc80 + j * 8;
      async_ld_b128(&Ks[buf][row * 64 + c8],
                    &Kb[(size_t)(bh * SEQ + kv0 + row) * HDIM + c8]);
      async_ld_b128(&Vt[buf][row * 64 + c8],
                    &Vtb[((size_t)bh * HDIM + row) * SEQ + kv0 + c8]);
    }
  };

  const int nkv = (q0 + 128) / 64;   // causal: only key blocks <= q block
  fill_kv(0, 0);
  wait_async0();
  __syncthreads();

  for (int kb = 0; kb < nkv; ++kb) {
    const int kv0 = kb * 64;
    const int cur = kb & 1;
    if (kb + 1 < nkv) fill_kv(kv0 + 64, cur ^ 1);

    // S = Q * K^T  (accumulate over d in two 32-chunks)
    v8f s[4];
#pragma unroll
    for (int ni = 0; ni < 4; ++ni) {
      v8f z = (v8f){};
#pragma unroll
      for (int kk = 0; kk < 2; ++kk) {
        v16bf bk = frag_split(Ks[cur], (ni * 16 + nIn) * 64 + kk * 32 + half * 8);
        z = __builtin_amdgcn_wmma_f32_16x16x32_bf16(
            false, aq[kk], false, bk, (short)0, z, false, false);
      }
      s[ni] = z;
    }

    unsigned short* Pw = Ps + wave * 16 * 64;
    // online softmax (rows live in 16-lane halves; xor-reduce within half)
#pragma unroll
    for (int r = 0; r < 8; ++r) {
      int qrow = qrow0 + half * 8 + r;
      float x[4];
      float mx = -1e30f;
#pragma unroll
      for (int ni = 0; ni < 4; ++ni) {
        int kn = kv0 + ni * 16 + nIn;
        float v = s[ni][r] * scale;
        v = (kn <= qrow) ? v : -1e30f;
        x[ni] = v;
        mx = fmaxf(mx, v);
      }
#pragma unroll
      for (int m2 = 8; m2 >= 1; m2 >>= 1) mx = fmaxf(mx, __shfl_xor(mx, m2, 32));
      float mnew = fmaxf(mrow[r], mx);
      float alpha = __expf(mrow[r] - mnew);
      mrow[r] = mnew;
      float rs = 0.f;
#pragma unroll
      for (int ni = 0; ni < 4; ++ni) {
        float p = __expf(x[ni] - mnew);
        rs += p;
        Pw[(half * 8 + r) * 64 + ni * 16 + nIn] = f2bf(p);
      }
#pragma unroll
      for (int m2 = 8; m2 >= 1; m2 >>= 1) rs += __shfl_xor(rs, m2, 32);
      lrow[r] = lrow[r] * alpha + rs;
#pragma unroll
      for (int ni = 0; ni < 4; ++ni) o[ni][r] *= alpha;
    }

    // O += P * V
#pragma unroll
    for (int kk = 0; kk < 2; ++kk) {
      v16bf ap = frag_split(Pw, nIn * 64 + kk * 32 + half * 8);
#pragma unroll
      for (int ni = 0; ni < 4; ++ni) {
        v16bf bv = frag_split(Vt[cur], (ni * 16 + nIn) * 64 + kk * 32 + half * 8);
        o[ni] = __builtin_amdgcn_wmma_f32_16x16x32_bf16(
            false, ap, false, bv, (short)0, o[ni], false, false);
      }
    }

    wait_async0();
    __syncthreads();
  }

  // normalize and write [B,T,EMBED] bf16 (head-concatenated)
  const int bidx = bh >> 4;
  const int h    = bh & 15;
#pragma unroll
  for (int r = 0; r < 8; ++r) {
    float inv = 1.0f / lrow[r];
    int t = qrow0 + half * 8 + r;
#pragma unroll
    for (int ni = 0; ni < 4; ++ni) {
      int d = ni * 16 + nIn;
      Ab[(size_t)(bidx * SEQ + t) * EMBED + h * HDIM + d] = f2bf(o[ni][r] * inv);
    }
  }
}

// ---------------------------------------------------------------------------
// Output projection: A[MTOT,EMBED] x Wt[EMBED,EMBED]^T + bias -> fp32 out
// ---------------------------------------------------------------------------
__global__ __launch_bounds__(256)
void proj_gemm_kernel(const unsigned short* __restrict__ Ab,
                      const unsigned short* __restrict__ WbT,  // [N][K]
                      const float* __restrict__ bias,
                      float* __restrict__ out) {
  __shared__ __align__(16) unsigned short As[2][128 * 64];
  __shared__ __align__(16) unsigned short Bs[2][128 * 64];
  const int tid  = threadIdx.x;
  const int lane = tid & 31;
  const int wave = tid >> 5;
  const int m0 = blockIdx.y * 128;
  const int n0 = blockIdx.x * 128;
  const int waveM = (wave >> 1) * 32;
  const int waveN = (wave & 1) * 64;
  const int nIn = lane & 15;
  const int half = lane >> 4;

  v8f acc[2][4];
#pragma unroll
  for (int mi = 0; mi < 2; ++mi)
#pragma unroll
    for (int ni = 0; ni < 4; ++ni) acc[mi][ni] = (v8f){};

  const int NIT = EMBED / 64;
  gemm_fill_async(As[0], Bs[0], Ab, WbT, m0, n0, 0, tid);
  wait_async0();
  __syncthreads();

  for (int it = 0; it < NIT; ++it) {
    const int cur = it & 1;
    if (it + 1 < NIT)
      gemm_fill_async(As[cur ^ 1], Bs[cur ^ 1], Ab, WbT, m0, n0, (it + 1) * 64, tid);

    const unsigned short* Ac = As[cur];
    const unsigned short* Bc = Bs[cur];
    v16bf af[2][2];
#pragma unroll
    for (int mi = 0; mi < 2; ++mi)
#pragma unroll
      for (int kk = 0; kk < 2; ++kk)
        af[mi][kk] = frag_split(Ac, (waveM + mi * 16 + nIn) * 64 + kk * 32 + half * 8);
#pragma unroll
    for (int ni = 0; ni < 4; ++ni) {
#pragma unroll
      for (int kk = 0; kk < 2; ++kk) {
        v16bf bf = frag_split(Bc, (waveN + ni * 16 + nIn) * 64 + kk * 32 + half * 8);
#pragma unroll
        for (int mi = 0; mi < 2; ++mi)
          acc[mi][ni] = __builtin_amdgcn_wmma_f32_16x16x32_bf16(
              false, af[mi][kk], false, bf, (short)0, acc[mi][ni], false, false);
      }
    }
    wait_async0();
    __syncthreads();
  }

#pragma unroll
  for (int mi = 0; mi < 2; ++mi)
#pragma unroll
    for (int ni = 0; ni < 4; ++ni) {
      int ncol = n0 + waveN + ni * 16 + nIn;
      float bv = bias[ncol];
#pragma unroll
      for (int r = 0; r < 8; ++r) {
        int mrow = m0 + waveM + mi * 16 + half * 8 + r;
        out[(size_t)mrow * EMBED + ncol] = acc[mi][ni][r] + bv;
      }
    }
}

// ---------------------------------------------------------------------------
extern "C" void kernel_launch(void* const* d_in, const int* in_sizes, int n_in,
                              void* d_out, int out_size, void* d_ws, size_t ws_size,
                              hipStream_t stream) {
  (void)in_sizes; (void)n_in; (void)out_size; (void)ws_size;
  const float* x      = (const float*)d_in[0];
  const float* W_qkv  = (const float*)d_in[1];
  const float* b_qkv  = (const float*)d_in[2];
  const float* W_proj = (const float*)d_in[3];
  const float* b_proj = (const float*)d_in[4];
  float* out = (float*)d_out;

  unsigned short* ws = (unsigned short*)d_ws;
  size_t off = 0;
  auto carve = [&](size_t elems) { unsigned short* p = ws + off; off += elems; return p; };
  unsigned short* xb    = carve((size_t)MTOT * EMBED);
  unsigned short* wqkvT = carve((size_t)EMBED * N_QKV);   // [N_QKV][EMBED] bf16
  unsigned short* wprjT = carve((size_t)EMBED * EMBED);   // [N][K] bf16
  unsigned short* Qb    = carve((size_t)MTOT * EMBED);    // [B,H,T,D]
  unsigned short* Kb    = carve((size_t)MTOT * EMBED);    // [B,H,T,D]
  unsigned short* Vtb   = carve((size_t)MTOT * EMBED);    // [B,H,D,T]
  unsigned short* Abuf  = carve((size_t)MTOT * EMBED);    // [B,T,EMBED]

  cvt_f32_bf16_kernel<<<512, 256, 0, stream>>>(x, xb, MTOT * EMBED);
  cvt_transpose_kernel<<<dim3(N_QKV / 32, EMBED / 32), 256, 0, stream>>>(
      W_qkv, wqkvT, EMBED, N_QKV);
  cvt_transpose_kernel<<<dim3(EMBED / 32, EMBED / 32), 256, 0, stream>>>(
      W_proj, wprjT, EMBED, EMBED);

  qkv_gemm_kernel<<<dim3(N_QKV / 128, MTOT / 128), 256, 0, stream>>>(
      xb, wqkvT, b_qkv, Qb, Kb, Vtb);

  attn_kernel<<<dim3(SEQ / 128, BATCH * HEADS), 256, 0, stream>>>(Qb, Kb, Vtb, Abuf);

  proj_gemm_kernel<<<dim3(EMBED / 128, MTOT / 128), 256, 0, stream>>>(
      Abuf, wprjT, b_proj, out);
}
